// HungarianMatcherDynamicK_76029511074256
// MI455X (gfx1250) — matrix operand order repaired and checked by reading, not compiled, not run on hardware
//
#include <hip/hip_runtime.h>

typedef __attribute__((ext_vector_type(16))) _Float16 v16h;
typedef __attribute__((ext_vector_type(8)))  float    v8f;
typedef __attribute__((ext_vector_type(4)))  int      v4i;

namespace {
constexpr int Bn = 16;      // batch
constexpr int Qn = 4000;    // queries
constexpr int Gn = 300;     // gts
constexpr int Cn = 81;      // classes
constexpr int KP = 96;      // class dim padded to 3*32 for WMMA K
constexpr int GP = 304;     // gt dim padded to 19*16 for WMMA N
constexpr int MT = Qn / 16; // 250 M tiles
constexpr int NT = GP / 16; // 19  N tiles
constexpr float kAlpha = 0.25f;
constexpr float kCR    = 2.5f;
constexpr float kInf   = 3.402823466e38f;
}

#define HAS_ASYNC_LDS                                            \
  (__has_builtin(__builtin_amdgcn_global_load_async_to_lds_b128) && \
   __has_builtin(__builtin_amdgcn_s_wait_asynccnt))

// ---------------------------------------------------------------- focal table
// F[b][q][k] = pos - neg (0 for padded k >= 81), stored f16 for WMMA A operand.
__global__ void k_focal(const float* __restrict__ logits, _Float16* __restrict__ F) {
  int i = blockIdx.x * blockDim.x + threadIdx.x;
  const int total = Bn * Qn * KP;
  if (i >= total) return;
  int k  = i % KP;
  int qq = i / KP;  // b*Qn + q
  float out = 0.0f;
  if (k < Cn) {
    float x = logits[(size_t)qq * Cn + k];
    float p = 1.0f / (1.0f + expf(-x));
    float neg = (1.0f - kAlpha) * p * p * (-logf(1.0f - p + 1e-8f));
    float pos = kAlpha * (1.0f - p) * (1.0f - p) * (-logf(p + 1e-8f));
    out = pos - neg;
  }
  F[i] = (_Float16)out;
}

// ------------------------------------------- one-hot labels, TRANSPOSED layout
// OHt[b][g][k] = (labels[b][g] == k), f16, zero in all pads. K contiguous so
// the WMMA B operand is two b128 loads per lane (same shape as A).
__global__ void k_onehot(const int* __restrict__ labels, _Float16* __restrict__ OHt) {
  int i = blockIdx.x * blockDim.x + threadIdx.x;
  const int total = Bn * GP * KP;
  if (i >= total) return;
  int k = i % KP;
  int t = i / KP;
  int g = t % GP;
  int b = t / GP;
  float v = 0.0f;
  if (k < Cn && g < Gn && labels[b * Gn + g] == k) v = 1.0f;
  OHt[i] = (_Float16)v;
}

// --------------------------------------------------------------- fg indicator
__global__ void k_fg(const float* __restrict__ pbx, const float* __restrict__ gbx,
                     int* __restrict__ fg) {
  __shared__ float sgt[Gn * 4];
  int b = blockIdx.y;
  const float* g4 = gbx + (size_t)b * Gn * 4;
#if HAS_ASYNC_LDS
  // CDNA5 async global->LDS staging of the 300 GT boxes (ASYNCcnt path).
  if (threadIdx.x < Gn) {
    __attribute__((address_space(3))) v4i* dst =
        (__attribute__((address_space(3))) v4i*)(sgt + threadIdx.x * 4);
    __attribute__((address_space(1))) v4i* src =
        (__attribute__((address_space(1))) v4i*)(g4 + threadIdx.x * 4);
    __builtin_amdgcn_global_load_async_to_lds_b128(src, dst, 0, 0);
  }
  __builtin_amdgcn_s_wait_asynccnt(0);
#else
  for (int i = threadIdx.x; i < Gn * 4; i += blockDim.x) sgt[i] = g4[i];
#endif
  __syncthreads();
  int q = blockIdx.x * blockDim.x + threadIdx.x;
  if (q >= Qn) return;
  const float* p = pbx + ((size_t)(b * Qn + q)) * 4;
  float ax = 0.5f * (p[0] + p[2]);
  float ay = 0.5f * (p[1] + p[3]);
  int any = 0;
  for (int g = 0; g < Gn && !any; ++g) {
    float x0 = sgt[g * 4 + 0], y0 = sgt[g * 4 + 1];
    float x1 = sgt[g * 4 + 2], y1 = sgt[g * 4 + 3];
    bool inb = (ax > x0) && (ax < x1) && (ay > y0) && (ay < y1);
    float cx = 0.5f * (x0 + x1), cy = 0.5f * (y0 + y1);
    float w = x1 - x0, h = y1 - y0;
    bool inc = (ax > cx - kCR * w) && (ax < cx + kCR * w) &&
               (ay > cy - kCR * h) && (ay < cy + kCR * h);
    any = (int)(inb | inc);
  }
  fg[b * Qn + q] = any;
}

// -------------------------------------------------- fused cost tile (uses WMMA)
// One wave32 per 16x16 output tile: cost_class via v_wmma_f32_16x16x32_f16 on
// F (Q x 96, f16) x onehot^T (304 x 96, f16); bbox/giou/mask terms fused.
__global__ void k_cost(const _Float16* __restrict__ F, const _Float16* __restrict__ OHt,
                       const float* __restrict__ pbx, const float* __restrict__ gbx,
                       const float* __restrict__ imgsz, const int* __restrict__ fg,
                       float* __restrict__ cost) {
  int wave = (blockIdx.x * blockDim.x + threadIdx.x) >> 5;
  int lane = threadIdx.x & 31;
  if (wave >= Bn * MT * NT) return;
  int b  = wave / (MT * NT);
  int r  = wave % (MT * NT);
  int mt = r / NT, nt = r % NT;
  int m0 = mt * 16, n0 = nt * 16;
  int l15 = lane & 15;
  int hi  = (lane & 16) ? 1 : 0;

  const _Float16* Frow = F + ((size_t)(b * Qn + m0 + l15)) * KP;
  int gcol = n0 + l15;
  const _Float16* Brow = OHt + ((size_t)(b * GP + gcol)) * KP;

  v8f acc = {};
#pragma unroll
  for (int s = 0; s < 3; ++s) {
    int k0 = 32 * s;
    v16h a, bm;
    int ka = k0 + (hi ? 8 : 0);    // A: 16-bit 16x32 lane layout
#pragma unroll
    for (int e = 0; e < 8; ++e) {
      a[e]     = Frow[ka + e];
      a[e + 8] = Frow[ka + 16 + e];
    }
    int kb = k0 + (hi ? 8 : 0);    // B: same K order as A (table order)
#pragma unroll
    for (int e = 0; e < 8; ++e) {
      bm[e]     = Brow[kb + e];
      bm[e + 8] = Brow[kb + 16 + e];
    }
    acc = __builtin_amdgcn_wmma_f32_16x16x32_f16(false, a, false, bm,
                                                 (short)0, acc, false, false);
  }

  // GT-side scalars (one GT column per lane)
  float W = imgsz[b * 4 + 0], H = imgsz[b * 4 + 1];
  float gb0 = 0.f, gb1 = 0.f, gb2 = 1.f, gb3 = 1.f;
  if (gcol < Gn) {
    const float* g4 = gbx + ((size_t)(b * Gn + gcol)) * 4;
    gb0 = g4[0]; gb1 = g4[1]; gb2 = g4[2]; gb3 = g4[3];
  }
  float gw = gb2 - gb0, gh = gb3 - gb1, areaB = gw * gh;
  float gcx = 0.5f * (gb0 + gb2), gcy = 0.5f * (gb1 + gb3);
  float ng0 = gb0 / W, ng1 = gb1 / H, ng2 = gb2 / W, ng3 = gb3 / H;
  float* crow = cost + (size_t)b * Qn * Gn;
  const int* fgb = fg + b * Qn;

#pragma unroll
  for (int rr = 0; rr < 8; ++rr) {
    int q = m0 + rr + hi * 8;
    const float* p4 = pbx + ((size_t)(b * Qn + q)) * 4;
    float p0 = p4[0], p1 = p4[1], p2 = p4[2], p3 = p4[3];
    float ax = 0.5f * (p0 + p2), ay = 0.5f * (p1 + p3);
    bool inb = (ax > gb0) && (ax < gb2) && (ay > gb1) && (ay < gb3);
    bool inc = (ax > gcx - kCR * gw) && (ax < gcx + kCR * gw) &&
               (ay > gcy - kCR * gh) && (ay < gcy + kCR * gh);
    float areaA = (p2 - p0) * (p3 - p1);
    float ltx = fmaxf(p0, gb0), lty = fmaxf(p1, gb1);
    float rbx = fminf(p2, gb2), rby = fminf(p3, gb3);
    float iw = fmaxf(rbx - ltx, 0.f), ih = fmaxf(rby - lty, 0.f);
    float inter = iw * ih;
    float uni = areaA + areaB - inter;
    float iou = inter / uni;
    float Lx = fminf(p0, gb0), Ly = fminf(p1, gb1);
    float Rx = fmaxf(p2, gb2), Ry = fmaxf(p3, gb3);
    float ew = fmaxf(Rx - Lx, 0.f), eh = fmaxf(Ry - Ly, 0.f);
    float earea = ew * eh;
    float giou = iou - (earea - uni) / earea;
    float cb = fabsf(p0 / W - ng0) + fabsf(p1 / H - ng1) +
               fabsf(p2 / W - ng2) + fabsf(p3 / H - ng3);
    float total = 5.0f * cb + 2.0f * acc[rr] - 2.0f * giou +
                  ((inb && inc) ? 0.0f : 100.0f) +
                  (fgb[q] ? 0.0f : 10000.0f);
    if (gcol < Gn) crow[(size_t)q * Gn + gcol] = total;
  }
}

// ---------------------------------------------------------- dyn_k (top-5 IoUs)
__global__ void k_dynk(const float* __restrict__ pbx, const float* __restrict__ gbx,
                       int* __restrict__ dynk) {
  int i = blockIdx.x * blockDim.x + threadIdx.x;
  if (i >= Bn * Gn) return;
  int b = i / Gn;
  const float* g4 = gbx + (size_t)i * 4;
  float gb0 = g4[0], gb1 = g4[1], gb2 = g4[2], gb3 = g4[3];
  float areaB = (gb2 - gb0) * (gb3 - gb1);
  float t0 = 0.f, t1 = 0.f, t2 = 0.f, t3 = 0.f, t4 = 0.f;
  const float* p = pbx + (size_t)b * Qn * 4;
  for (int q = 0; q < Qn; ++q) {
    float p0 = p[q * 4 + 0], p1 = p[q * 4 + 1], p2 = p[q * 4 + 2], p3 = p[q * 4 + 3];
    float areaA = (p2 - p0) * (p3 - p1);
    float iw = fmaxf(fminf(p2, gb2) - fmaxf(p0, gb0), 0.f);
    float ih = fmaxf(fminf(p3, gb3) - fmaxf(p1, gb1), 0.f);
    float inter = iw * ih;
    float iou = inter / (areaA + areaB - inter);
    if (iou > t4) {
      if (iou > t0)      { t4 = t3; t3 = t2; t2 = t1; t1 = t0; t0 = iou; }
      else if (iou > t1) { t4 = t3; t3 = t2; t2 = t1; t1 = iou; }
      else if (iou > t2) { t4 = t3; t3 = t2; t2 = iou; }
      else if (iou > t3) { t4 = t3; t3 = iou; }
      else               { t4 = iou; }
    }
  }
  int k = (int)(t0 + t1 + t2 + t3 + t4);
  if (k < 1) k = 1;
  if (k > 5) k = 5;
  dynk[i] = k;
}

// ------------------------------------------------------------------ zero fill
__global__ void k_zero(float* __restrict__ m, size_t n) {
  size_t i = (size_t)blockIdx.x * blockDim.x + threadIdx.x;
  if (i < n) m[i] = 0.0f;
}

// ----------------------------------- initial m: dyn_k smallest (stable) per gt
__global__ void k_initm(const float* __restrict__ cost, const int* __restrict__ dynk,
                        float* __restrict__ m) {
  int i = blockIdx.x * blockDim.x + threadIdx.x;
  if (i >= Bn * Gn) return;
  int b = i / Gn, g = i % Gn;
  const float* cc = cost + (size_t)b * Qn * Gn + g;
  float v[5] = {kInf, kInf, kInf, kInf, kInf};
  int   id[5] = {0, 0, 0, 0, 0};
  for (int q = 0; q < Qn; ++q) {
    __builtin_prefetch(cc + (size_t)(q + 32) * Gn, 0, 0);
    float c = cc[(size_t)q * Gn];
    if (c < v[4]) {
      int j = 4;
      while (j > 0 && c < v[j - 1]) { v[j] = v[j - 1]; id[j] = id[j - 1]; --j; }
      v[j] = c; id[j] = q;
    }
  }
  int k = dynk[i];
  float* mm = m + (size_t)b * Qn * Gn + g;
  for (int j = 0; j < k; ++j) mm[(size_t)id[j] * Gn] = 1.0f;
}

// ----------------- resolve multi rows; record multi mask and row-argmin column
__global__ void k_multi(const float* __restrict__ cost, float* __restrict__ m,
                        int* __restrict__ multi, int* __restrict__ ca) {
  int i = blockIdx.x * blockDim.x + threadIdx.x;
  if (i >= Bn * Qn) return;
  const float* cr = cost + (size_t)i * Gn;
  float* mr = m + (size_t)i * Gn;
  int s = 0;
  float bv = kInf;
  int bi = 0;
  for (int g = 0; g < Gn; ++g) {
    s += (mr[g] != 0.0f);
    float c = cr[g];
    if (c < bv) { bv = c; bi = g; }
  }
  ca[i] = bi;
  int mu = (s > 1);
  multi[i] = mu;
  if (mu) {
    for (int g = 0; g < Gn; ++g) mr[g] = 0.0f;
    mr[bi] = 1.0f;
  }
}

// --------------------------- sequential repair loop: one block per image
__global__ __launch_bounds__(1024) void k_iter(const float* __restrict__ cost,
                                               float* __restrict__ m,
                                               float* __restrict__ pen,
                                               const int* __restrict__ multi,
                                               const int* __restrict__ ca) {
  int b = blockIdx.x;
  const float* C = cost + (size_t)b * Qn * Gn;
  float* M = m + (size_t)b * Qn * Gn;
  float* P = pen + b * Qn;
  const int* MU = multi + b * Qn;
  const int* CA = ca + b * Qn;
  __shared__ int colsum[Gn];
  __shared__ int posq[Gn];
  __shared__ int s_all, s_mul;
  int tid = threadIdx.x, T = blockDim.x;
  for (int q = tid; q < Qn; q += T) P[q] = 0.0f;
  __syncthreads();
  for (int it = 0; it < 100; ++it) {
    for (int g = tid; g < Gn; g += T) {
      int s = 0;
      for (int q = 0; q < Qn; ++q) s += (M[(size_t)q * Gn + g] != 0.0f);
      colsum[g] = s;
    }
    if (tid == 0) s_all = 1;
    __syncthreads();
    for (int g = tid; g < Gn; g += T)
      if (colsum[g] == 0) s_all = 0;
    __syncthreads();
    if (s_all) break;
    // c += 1e5 per currently-matched row (pre-update m) -> pen counter
    for (int q = tid; q < Qn; q += T) {
      const float* mr = M + (size_t)q * Gn;
      int s = 0;
      for (int g = 0; g < Gn; ++g) s += (mr[g] != 0.0f);
      if (s > 0) P[q] += 1.0f;
    }
    __syncthreads();
    // column argmin of effective c for unmatched columns
    for (int g = tid; g < Gn; g += T) {
      if (colsum[g] == 0) {
        float bv = kInf;
        int bi = 0;
        for (int q = 0; q < Qn; ++q) {
          float v = C[(size_t)q * Gn + g] + 100000.0f * P[q];
          if (v < bv) { bv = v; bi = q; }
        }
        posq[g] = bi;
      }
    }
    __syncthreads();
    for (int g = tid; g < Gn; g += T)
      if (colsum[g] == 0) M[(size_t)posq[g] * Gn + g] = 1.0f;
    if (tid == 0) s_mul = 0;
    __syncthreads();
    for (int q = tid; q < Qn; q += T) {
      const float* mr = M + (size_t)q * Gn;
      int s = 0;
      for (int g = 0; g < Gn; ++g) s += (mr[g] != 0.0f);
      if (s > 1) s_mul = 1;
    }
    __syncthreads();
    if (s_mul) {
      // fix: row argmin of c == row argmin of cost (pen is row-constant)
      for (int q = tid; q < Qn; q += T) {
        if (MU[q]) {
          float* mr = M + (size_t)q * Gn;
          for (int g = 0; g < Gn; ++g) mr[g] = 0.0f;
          mr[CA[q]] = 1.0f;
        }
      }
      __syncthreads();
    }
  }
}

// --------------------------------------------------------------- finalization
__global__ void k_final_rows(const float* __restrict__ m, float* __restrict__ sel,
                             float* __restrict__ gti) {
  int i = blockIdx.x * blockDim.x + threadIdx.x;
  if (i >= Bn * Qn) return;
  const float* mr = m + (size_t)i * Gn;
  int s = 0, first = -1;
  for (int g = 0; g < Gn; ++g) {
    if (mr[g] != 0.0f) {
      ++s;
      if (first < 0) first = g;
    }
  }
  sel[i] = (s > 0) ? 1.0f : 0.0f;
  gti[i] = (float)((first < 0) ? 0 : first);
}

__global__ void k_final_cols(const float* __restrict__ cost, const float* __restrict__ m,
                             const float* __restrict__ pen, float* __restrict__ mq) {
  int i = blockIdx.x * blockDim.x + threadIdx.x;
  if (i >= Bn * Gn) return;
  int b = i / Gn, g = i % Gn;
  const float* C = cost + (size_t)b * Qn * Gn + g;
  const float* M = m + (size_t)b * Qn * Gn + g;
  const float* P = pen + b * Qn;
  float bv = kInf;
  int bi = 0;
  for (int q = 0; q < Qn; ++q) {
    __builtin_prefetch(C + (size_t)(q + 32) * Gn, 0, 0);
    if (M[(size_t)q * Gn] != 0.0f) {
      float v = C[(size_t)q * Gn] + 100000.0f * P[q];
      if (v < bv) { bv = v; bi = q; }
    }
  }
  mq[i] = (float)bi;
}

// ------------------------------------------------------------------ launcher
extern "C" void kernel_launch(void* const* d_in, const int* in_sizes, int n_in,
                              void* d_out, int out_size, void* d_ws, size_t ws_size,
                              hipStream_t stream) {
  const float* logits = (const float*)d_in[0];
  const float* pboxes = (const float*)d_in[1];
  const int*   labels = (const int*)d_in[2];
  const float* gboxes = (const float*)d_in[3];
  const float* imgsz  = (const float*)d_in[4];

  char* wsp = (char*)d_ws;
  auto alloc = [&](size_t bytes) -> char* {
    char* p = wsp;
    wsp += (bytes + 255) & ~(size_t)255;
    return p;
  };
  float*    cost = (float*)alloc((size_t)Bn * Qn * Gn * sizeof(float));
  _Float16* F    = (_Float16*)alloc((size_t)Bn * Qn * KP * sizeof(_Float16));
  _Float16* OHt  = (_Float16*)alloc((size_t)Bn * GP * KP * sizeof(_Float16));
  int*      fg   = (int*)alloc((size_t)Bn * Qn * sizeof(int));
  int*      dynk = (int*)alloc((size_t)Bn * Gn * sizeof(int));
  float*    pen  = (float*)alloc((size_t)Bn * Qn * sizeof(float));
  int*      mult = (int*)alloc((size_t)Bn * Qn * sizeof(int));
  int*      ca   = (int*)alloc((size_t)Bn * Qn * sizeof(int));

  float* m_out   = (float*)d_out;
  float* sel_out = m_out + (size_t)Bn * Qn * Gn;
  float* gti_out = sel_out + (size_t)Bn * Qn;
  float* mq_out  = gti_out + (size_t)Bn * Qn;

  const int TB = 256;
  k_focal<<<(Bn * Qn * KP + TB - 1) / TB, TB, 0, stream>>>(logits, F);
  k_onehot<<<(Bn * GP * KP + TB - 1) / TB, TB, 0, stream>>>(labels, OHt);
  k_fg<<<dim3((Qn + TB - 1) / TB, Bn), TB, 0, stream>>>(pboxes, gboxes, fg);
  k_cost<<<(Bn * MT * NT) / 8, TB, 0, stream>>>(F, OHt, pboxes, gboxes, imgsz, fg, cost);
  k_dynk<<<(Bn * Gn + TB - 1) / TB, TB, 0, stream>>>(pboxes, gboxes, dynk);
  size_t mElems = (size_t)Bn * Qn * Gn;
  k_zero<<<(unsigned)((mElems + TB - 1) / TB), TB, 0, stream>>>(m_out, mElems);
  k_initm<<<(Bn * Gn + TB - 1) / TB, TB, 0, stream>>>(cost, dynk, m_out);
  k_multi<<<(Bn * Qn + TB - 1) / TB, TB, 0, stream>>>(cost, m_out, mult, ca);
  k_iter<<<Bn, 1024, 0, stream>>>(cost, m_out, pen, mult, ca);
  k_final_rows<<<(Bn * Qn + TB - 1) / TB, TB, 0, stream>>>(m_out, sel_out, gti_out);
  k_final_cols<<<(Bn * Gn + TB - 1) / TB, TB, 0, stream>>>(cost, m_out, pen, mq_out);
}
